// Encoder_58823872086604
// MI455X (gfx1250) — matrix operand order repaired and checked by reading, not compiled, not run on hardware
//
#include <hip/hip_runtime.h>
#include <hip/hip_bf16.h>

// ---------------- problem constants ----------------
#define B_   32
#define T_   1024
#define D_   512
#define H_   512
#define KTOT 1024      // D + H (fused [x;h] contraction dim)
#define NWG  32        // workgroups per direction
#define HC   16        // h-columns owned per WG
#define NCOL 64        // gate columns per WG (4 gates x HC)
#define NTHREADS 256   // 8 waves (wave32)

#ifndef __has_builtin
#define __has_builtin(x) 0
#endif
#if __has_builtin(__builtin_amdgcn_global_load_async_to_lds_b128)
#define HAVE_ASYNC_B128 1
#else
#define HAVE_ASYNC_B128 0
#endif
#if __has_builtin(__builtin_amdgcn_tensor_load_to_lds)
#define HAVE_TDM 1
#else
#define HAVE_TDM 0
#endif

typedef __attribute__((ext_vector_type(16))) __bf16 v16bf;
typedef __attribute__((ext_vector_type(8)))  float  v8f;
typedef __attribute__((ext_vector_type(4)))  unsigned int u32x4;
typedef __attribute__((ext_vector_type(8)))  int i32x8;
typedef __attribute__((ext_vector_type(4)))  int i32x4;

// exact pointer types the async-LDS builtins expect ("vV4i*1V4i*3IiIi")
#define GLOBAL_V4(p) ((__attribute__((address_space(1))) i32x4*)(p))
#define LDS_V4(p)    ((__attribute__((address_space(3))) i32x4*)(p))

union AF { uint4 q[2]; v16bf v; };   // 32B fragment = 16 bf16 per lane

__device__ __forceinline__ unsigned short f2bf(float f) {
  unsigned u = __float_as_uint(f);
  unsigned r = u + 0x7FFFu + ((u >> 16) & 1u);   // round-to-nearest-even
  return (unsigned short)(r >> 16);
}
__device__ __forceinline__ float sigf(float x) { return 1.0f / (1.0f + __expf(-x)); }

__device__ __forceinline__ void wait_async0() {
#if __has_builtin(__builtin_amdgcn_s_wait_asynccnt)
  __builtin_amdgcn_s_wait_asynccnt(0);
#else
  asm volatile("s_wait_asynccnt 0" ::: "memory");
#endif
}
__device__ __forceinline__ void wait_tensor0() {
#if __has_builtin(__builtin_amdgcn_s_wait_tensorcnt)
  __builtin_amdgcn_s_wait_tensorcnt(0);
#else
  asm volatile("s_wait_tensorcnt 0" ::: "memory");
#endif
}

// 16B copy to LDS: async (no VGPR round-trip, ASYNCcnt) when available
__device__ __forceinline__ void copy16_to_lds(unsigned short* dst,
                                              const unsigned short* src) {
#if HAVE_ASYNC_B128
  __builtin_amdgcn_global_load_async_to_lds_b128(GLOBAL_V4(src), LDS_V4(dst), 0, 0);
#else
  *(uint4*)dst = *(const uint4*)src;
#endif
}

// ---------------- small helper kernels ----------------
__global__ void cvt_f32_bf16(const float* __restrict__ src,
                             unsigned short* __restrict__ dst, size_t n) {
  size_t i = (size_t)blockIdx.x * blockDim.x + threadIdx.x;
  size_t stride = (size_t)gridDim.x * blockDim.x;
  for (; i < n; i += stride) dst[i] = f2bf(src[i]);
}

// wpack layout: [dir][wg][col][k] bf16 ; col = gate*16+n ; k<512 from Wk, else Wr
__global__ void pack_w_kernel(const float* __restrict__ Wk,
                              const float* __restrict__ Wr,
                              unsigned short* __restrict__ wp, int layer) {
  size_t i = (size_t)blockIdx.x * blockDim.x + threadIdx.x;   // 2*32*64*1024 elems
  if (i >= (size_t)2 * NWG * NCOL * KTOT) return;
  int k   = (int)(i & 1023);
  size_t ci = i >> 10;
  int col = (int)(ci & 63);
  size_t wi = ci >> 6;
  int wg  = (int)(wi & 31);
  int dir = (int)(wi >> 5);
  int gate = col >> 4, n = col & 15;
  int gcol = gate * H_ + wg * HC + n;            // column in 4H=2048
  size_t base = ((size_t)layer * 2 + dir) * 512 * 2048;
  float v = (k < 512) ? Wk[base + (size_t)k * 2048 + gcol]
                      : Wr[base + (size_t)(k - 512) * 2048 + gcol];
  wp[i] = f2bf(v);
}

__global__ void add_kernel(const float* __restrict__ a, const float* __restrict__ b,
                           float* __restrict__ o, size_t n) {
  size_t i = (size_t)blockIdx.x * blockDim.x + threadIdx.x;
  size_t stride = (size_t)gridDim.x * blockDim.x;
  for (; i < n; i += stride) o[i] = a[i] + b[i];
}

// ---------------- persistent bidirectional LSTM layer ----------------
__global__ void __launch_bounds__(NTHREADS, 1)
lstm_layer_kernel(const unsigned short* __restrict__ xbf,   // [B][T][D] bf16 layer input
                  const unsigned char*  __restrict__ mask,  // [B][T]
                  const unsigned short* __restrict__ wpack, // [2][NWG][NCOL][KTOT] bf16
                  const float*          __restrict__ bias,  // this layer: [2][2048]
                  unsigned short*       __restrict__ hbf,   // [2][B][H] bf16 broadcast state
                  float*                __restrict__ y,     // [B][T][H] fwd+bwd accumulated
                  int*                  __restrict__ cnt)   // [2] barrier counters (zeroed)
{
  extern __shared__ __align__(16) char smem[];
  unsigned short* w_sh = (unsigned short*)smem;          // 64*1024 bf16 = 128 KB
  unsigned short* a_sh = w_sh + NCOL * KTOT;             // 32*1024 bf16 =  64 KB
  float* z_sh = (float*)(a_sh + B_ * KTOT);              // [4][32][16] f32 = 8 KB
  float* c_sh = z_sh + 4 * B_ * HC;                      // [32][16]
  float* h_sh = c_sh + B_ * HC;                          // [32][16]
  float* b_sh = h_sh + B_ * HC;                          // [64]

  const int tid   = threadIdx.x;
  const int dir   = blockIdx.x >> 5;     // 0=fwd, 1=bwd
  const int wg    = blockIdx.x & 31;     // h-column chunk
  const int lane  = tid & 31;
  const int wid   = tid >> 5;            // 8 waves
  const int mtile = wid >> 2;            // 0..1 (M = 32 -> two 16-row tiles)
  const int gate  = wid & 3;             // wave owns one gate's 16 columns
  const int half  = lane >> 4;           // ISA lane<16 / lane>=16 K split
  const int l16   = lane & 15;

  // -------- stage this WG's weight slice into LDS (once per layer) --------
  {
    const unsigned short* wsrc = wpack + ((size_t)(dir * NWG + wg)) * NCOL * KTOT;
#if HAVE_TDM
    if (wid == 0) {
      // 1-D TDM copy: 128 KB as 16384 x 8-byte elements (D# per CDNA5 ISA)
      unsigned ldsoff = (unsigned)(unsigned long long)
          (__attribute__((address_space(3))) char*)(void*)w_sh;
      unsigned long long ga = (unsigned long long)wsrc;
      u32x4 g0;
      g0[0] = 1u;                                  // count=1, user mode
      g0[1] = ldsoff;                              // lds_addr (bytes)
      g0[2] = (unsigned)(ga & 0xFFFFFFFFu);        // global_addr[31:0]
      g0[3] = (unsigned)((ga >> 32) & 0x01FFFFFFu) // global_addr[56:32]
              | 0x80000000u;                       // type=2 ("image")
      i32x8 g1;
      g1[0] = 0x00030000;   // workgroup_mask=0, data_size=3 (8B)
      g1[1] = 0x40000000;   // tensor_dim0[15:0]=16384 (bits 63:48)
      g1[2] = 0x00010000;   // tensor_dim1=1 (bits 95:80)
      g1[3] = 0x40000000;   // tile_dim0=16384 (bits 127:112)
      g1[4] = 0x00000001;   // tile_dim1=1
      g1[5] = 0x00004000;   // tensor_dim0_stride=16384
      g1[6] = 0; g1[7] = 0;
      i32x4 z4 = {0, 0, 0, 0};
#if __clang_major__ >= 23
      i32x8 z8 = {0, 0, 0, 0, 0, 0, 0, 0};
      __builtin_amdgcn_tensor_load_to_lds(g0, g1, z4, z4, z8, 0);
#else
      __builtin_amdgcn_tensor_load_to_lds(g0, g1, z4, z4, 0);
#endif
      wait_tensor0();
    }
#else
    for (int ch = tid; ch < NCOL * KTOT / 8; ch += NTHREADS) {
      int col = ch >> 7; int ko = (ch & 127) * 8;
      copy16_to_lds(w_sh + col * KTOT + ko, wsrc + col * KTOT + ko);
    }
#if HAVE_ASYNC_B128
    wait_async0();
#endif
#endif
  }
  if (tid < NCOL) {
    int g = tid >> 4, n = tid & 15;
    b_sh[tid] = bias[dir * 4 * H_ + g * H_ + wg * HC + n];
  }
  for (int i = tid; i < B_ * HC; i += NTHREADS) { c_sh[i] = 0.0f; h_sh[i] = 0.0f; }
  __syncthreads();

  for (int s = 0; s < T_; ++s) {
    const int t = (dir == 0) ? s : (T_ - 1 - s);

    // -------- stage A = [x_t ; h] (32 x 1024 bf16) via async global->LDS --------
    for (int ch = tid; ch < B_ * KTOT / 8; ch += NTHREADS) {
      int row = ch >> 7; int ko = (ch & 127) * 8;
      const unsigned short* src =
          (ko < D_) ? xbf + (((size_t)row) * T_ + t) * D_ + ko
                    : hbf + (size_t)dir * B_ * H_ + row * H_ + (ko - D_);
      copy16_to_lds(a_sh + row * KTOT + ko, src);
    }
#if HAVE_ASYNC_B128
    wait_async0();
#endif
    // prefetch next timestep's x rows into cache
    if (s + 1 < T_ && tid < B_) {
      int tn = (dir == 0) ? (t + 1) : (t - 1);
      __builtin_prefetch(xbf + (((size_t)tid) * T_ + tn) * D_, 0, 1);
    }
    __syncthreads();

    // -------- (32x1024)@(1024x64) via v_wmma_f32_16x16x32_bf16 --------
    // Ping-pong double buffering: loads for step k+1 issued before WMMA of k
    v8f acc = {};
    const unsigned short* aRow = a_sh + (mtile * 16 + l16) * KTOT;  // A: row per lane
    const unsigned short* bCol = w_sh + (gate * 16 + l16) * KTOT;   // B: column per lane
#define LOADFRAG(Af, Bf, kb)                                          \
    Af.q[0] = *(const uint4*)(aRow + (kb) + half * 8);                \
    Af.q[1] = *(const uint4*)(aRow + (kb) + 16 + half * 8);           \
    Bf.q[0] = *(const uint4*)(bCol + (kb) + half * 8);                \
    Bf.q[1] = *(const uint4*)(bCol + (kb) + 16 + half * 8);
    AF a0, b0, a1, b1;
    LOADFRAG(a0, b0, 0)
    #pragma unroll
    for (int kk = 0; kk < 30; kk += 2) {
      LOADFRAG(a1, b1, (kk + 1) * 32)
      acc = __builtin_amdgcn_wmma_f32_16x16x32_bf16(false, a0.v, false, b0.v,
                                                    (short)0, acc, false, false);
      LOADFRAG(a0, b0, (kk + 2) * 32)
      acc = __builtin_amdgcn_wmma_f32_16x16x32_bf16(false, a1.v, false, b1.v,
                                                    (short)0, acc, false, false);
    }
    LOADFRAG(a1, b1, 31 * 32)
    acc = __builtin_amdgcn_wmma_f32_16x16x32_bf16(false, a0.v, false, b0.v,
                                                  (short)0, acc, false, false);
    acc = __builtin_amdgcn_wmma_f32_16x16x32_bf16(false, a1.v, false, b1.v,
                                                  (short)0, acc, false, false);
#undef LOADFRAG

    // spill z tile (+bias); C layout: VGPR r -> M = r + 8*(lane/16), N = lane%16
    {
      float bn = b_sh[gate * 16 + l16];
      #pragma unroll
      for (int r = 0; r < 8; ++r) {
        int m = mtile * 16 + half * 8 + r;
        z_sh[(gate * B_ + m) * HC + l16] = acc[r] + bn;
      }
    }
    __syncthreads();

    // -------- pointwise LSTM cell (f32) with mask carry --------
    for (int idx = tid; idx < B_ * HC; idx += NTHREADS) {
      int m = idx >> 4, n = idx & 15;
      float zi = z_sh[(0 * B_ + m) * HC + n];
      float zf = z_sh[(1 * B_ + m) * HC + n];
      float zg = z_sh[(2 * B_ + m) * HC + n];
      float zo = z_sh[(3 * B_ + m) * HC + n];
      float co = c_sh[idx];
      float cn = sigf(zf) * co + sigf(zi) * tanhf(zg);
      float hn = sigf(zo) * tanhf(cn);
      bool  mk = mask[(size_t)m * T_ + t] != 0;
      float cs = mk ? cn : co;
      float hs = mk ? hn : h_sh[idx];
      c_sh[idx] = cs;
      h_sh[idx] = hs;
      atomicAdd(&y[((size_t)m * T_ + t) * H_ + wg * HC + n], hs);  // fwd+bwd fuse
      hbf[(size_t)dir * B_ * H_ + m * H_ + wg * HC + n] = f2bf(hs);
    }

    // -------- per-direction grid barrier (release -> arrive -> spin -> acquire) --------
    __threadfence();
    __syncthreads();
    if (tid == 0) {
      __hip_atomic_fetch_add(&cnt[dir], 1, __ATOMIC_RELEASE, __HIP_MEMORY_SCOPE_AGENT);
      const int target = NWG * (s + 1);
      while (__hip_atomic_load(&cnt[dir], __ATOMIC_RELAXED, __HIP_MEMORY_SCOPE_AGENT) < target)
        __builtin_amdgcn_s_sleep(1);
    }
    __syncthreads();
    __builtin_amdgcn_fence(__ATOMIC_ACQUIRE, "agent");   // invalidate stale h in near caches
  }
}

// ---------------- host launcher ----------------
extern "C" void kernel_launch(void* const* d_in, const int* in_sizes, int n_in,
                              void* d_out, int out_size, void* d_ws, size_t ws_size,
                              hipStream_t stream) {
  const float*         x    = (const float*)d_in[0];
  const unsigned char* mask = (const unsigned char*)d_in[1];   // jnp bool -> 1 byte
  const float*         Wk   = (const float*)d_in[2];
  const float*         Wr   = (const float*)d_in[3];
  const float*         bia  = (const float*)d_in[4];
  float* out = (float*)d_out;

  char* ws = (char*)d_ws;
  float*          y0    = (float*)(ws + 0);                    // 64 MB
  float*          y1    = (float*)(ws + 67108864);             // 64 MB
  unsigned short* wpack = (unsigned short*)(ws + 134217728);   // 8 MB
  unsigned short* xbf   = (unsigned short*)(ws + 142606336);   // 32 MB
  unsigned short* hbf   = (unsigned short*)(ws + 176160768);   // 64 KB
  int*            cnt   = (int*)(ws + 176226304);              // 2 ints (right after hbf)

  const size_t SMEM = (size_t)(NCOL * KTOT + B_ * KTOT) * 2 +
                      (size_t)(4 * B_ * HC + 2 * B_ * HC + NCOL) * 4;   // ~204 KB
  (void)hipFuncSetAttribute(reinterpret_cast<const void*>(lstm_layer_kernel),
                            hipFuncAttributeMaxDynamicSharedMemorySize, (int)SMEM);

  const size_t NX = (size_t)B_ * T_ * D_;   // = B*T*H too
  const int PW_BLOCKS = (2 * NWG * NCOL * KTOT + 255) / 256;

  // ---- layer 0 ----
  cvt_f32_bf16<<<8192, 256, 0, stream>>>(x, xbf, NX);
  pack_w_kernel<<<PW_BLOCKS, 256, 0, stream>>>(Wk, Wr, wpack, 0);
  (void)hipMemsetAsync(y0, 0, NX * sizeof(float), stream);
  (void)hipMemsetAsync(hbf, 0, (size_t)2 * B_ * H_ * 2 + 64, stream);   // hbf + cnt
  lstm_layer_kernel<<<2 * NWG, NTHREADS, SMEM, stream>>>(xbf, mask, wpack,
                                                         bia, hbf, y0, cnt);
  // ---- layer 1 (input = y0, residual handled in final add) ----
  cvt_f32_bf16<<<8192, 256, 0, stream>>>(y0, xbf, NX);
  pack_w_kernel<<<PW_BLOCKS, 256, 0, stream>>>(Wk, Wr, wpack, 1);
  (void)hipMemsetAsync(y1, 0, NX * sizeof(float), stream);
  (void)hipMemsetAsync(hbf, 0, (size_t)2 * B_ * H_ * 2 + 64, stream);
  lstm_layer_kernel<<<2 * NWG, NTHREADS, SMEM, stream>>>(xbf, mask, wpack,
                                                         bia + 4 * H_, hbf, y1, cnt);
  // ---- out = (fwd1+bwd1) + (fwd0+bwd0) ----
  add_kernel<<<8192, 256, 0, stream>>>(y0, y1, out, NX);
}
